// MixPathGraphConvolution_50113678409792
// MI455X (gfx1250) — matrix operand order repaired and checked by reading, not compiled, not run on hardware
//
#include <hip/hip_runtime.h>

#define N_NODES 100000
#define N_EDGES 3200000
#define FEAT 256
#define HID 128
#define OUTC 64
#define HOPS 3

typedef __attribute__((ext_vector_type(2)))  float  v2f;
typedef __attribute__((ext_vector_type(8)))  float  v8f;
typedef __attribute__((ext_vector_type(16))) __bf16 v16bf;

// Exact split: x = hi + lo with hi = truncate-to-bf16(x) (exactly representable),
// lo = RNE-bf16(x - hi). Product ah*bh + ah*bl + al*bh ~ f32 product (err ~2^-16).
static __device__ __forceinline__ void split_f32(float x, __bf16& hi, __bf16& lo) {
    const unsigned u   = __builtin_bit_cast(unsigned, x);
    hi = __builtin_bit_cast(__bf16, (unsigned short)(u >> 16));
    const float hif = __builtin_bit_cast(float, u & 0xffff0000u);
    lo = (__bf16)(x - hif);
}

// ---------------------------------------------------------------------------
// x = A @ W + b ; feat = w0 * x        (bf16 split-3 path, f32 accumulate)
// A: [N, FEAT], W: [FEAT, HID], b: [HID]
// Block = 256 threads (8 waves); wave = 32 rows x 128 cols
// (2 row tiles x 8 col tiles of V_WMMA_F32_16X16X32_BF16, 3 products each).
// LDS: W hi/lo bf16 panels transposed as [col][K]  (2 x 64 KB).
// ---------------------------------------------------------------------------
__global__ void __launch_bounds__(256)
mpgc_gemm_in_kernel(const float* __restrict__ A,
                    const float* __restrict__ W,
                    const float* __restrict__ bias,
                    const float* __restrict__ w0p,
                    float* __restrict__ xout,
                    float* __restrict__ featout)
{
    extern __shared__ __bf16 ldsB[];
    __bf16* Whi = ldsB;                  // [HID][FEAT]
    __bf16* Wlo = ldsB + HID * FEAT;     // [HID][FEAT]

    const int tid = threadIdx.x;
    for (int i = tid; i < FEAT * HID; i += 256) {
        const int k = i >> 7;            // i / HID
        const int c = i & (HID - 1);     // i % HID
        __bf16 h, l;
        split_f32(W[i], h, l);
        Whi[c * FEAT + k] = h;
        Wlo[c * FEAT + k] = l;
    }
    __syncthreads();

    const int lane    = tid & 31;
    const int wave    = tid >> 5;
    const int rowBase = blockIdx.x * 256 + wave * 32;
    const int m       = lane & 15;
    const int koff    = (lane >> 4) << 4;   // A frag: 0 / 16
    const int kboff   = (lane >> 4) << 4;   // B frag: 0 / 16

    const float* Arow[2];
#pragma unroll
    for (int rt = 0; rt < 2; ++rt) {
        int r = rowBase + rt * 16 + m;
        if (r >= N_NODES) r = N_NODES - 1;   // clamp loads; stores predicated
        Arow[rt] = A + (size_t)r * FEAT;
    }

    v8f acc[2][8];
#pragma unroll
    for (int rt = 0; rt < 2; ++rt)
#pragma unroll
        for (int ct = 0; ct < 8; ++ct)
#pragma unroll
            for (int r = 0; r < 8; ++r) acc[rt][ct][r] = 0.0f;

    for (int kb = 0; kb < FEAT; kb += 32) {
        // --- A fragments: two contiguous 8-float runs per lane, split to hi/lo
        v16bf ah[2], al[2];
#pragma unroll
        for (int rt = 0; rt < 2; ++rt) {
            const v8f r0 = *(const v8f*)(Arow[rt] + kb + koff);
            const v8f r1 = *(const v8f*)(Arow[rt] + kb + 8 + koff);
#pragma unroll
            for (int e = 0; e < 8; ++e) {
                __bf16 h, l;
                split_f32(r0[e], h, l); ah[rt][e]     = h; al[rt][e]     = l;
                split_f32(r1[e], h, l); ah[rt][8 + e] = h; al[rt][8 + e] = l;
            }
        }
        // --- per column tile: batch B loads, then 6 back-to-back WMMAs
#pragma unroll
        for (int ct = 0; ct < 8; ++ct) {
            const int col = ct * 16 + m;
            const v16bf bh = *(const v16bf*)(Whi + (size_t)col * FEAT + kb + kboff);
            const v16bf bl = *(const v16bf*)(Wlo + (size_t)col * FEAT + kb + kboff);
#pragma unroll
            for (int rt = 0; rt < 2; ++rt) {
                acc[rt][ct] = __builtin_amdgcn_wmma_f32_16x16x32_bf16(
                    false, ah[rt], false, bh, (short)0, acc[rt][ct], false, false);
                acc[rt][ct] = __builtin_amdgcn_wmma_f32_16x16x32_bf16(
                    false, ah[rt], false, bl, (short)0, acc[rt][ct], false, false);
                acc[rt][ct] = __builtin_amdgcn_wmma_f32_16x16x32_bf16(
                    false, al[rt], false, bh, (short)0, acc[rt][ct], false, false);
            }
        }
    }

    const float w0  = w0p[0];
    const int   hi8 = (lane >> 4) << 3;   // C/D layout: lanes 16-31 hold M = r+8
#pragma unroll
    for (int rt = 0; rt < 2; ++rt) {
#pragma unroll
        for (int ct = 0; ct < 8; ++ct) {
            const int   col = ct * 16 + m;
            const float bv  = bias[col];
#pragma unroll
            for (int r = 0; r < 8; ++r) {
                const int row = rowBase + rt * 16 + r + hi8;
                if (row < N_NODES) {
                    const float v = acc[rt][ct][r] + bv;
                    xout[(size_t)row * HID + col]    = v;
                    featout[(size_t)row * HID + col] = w0 * v;
                }
            }
        }
    }
}

// ---------------------------------------------------------------------------
// out = [featS | featT] @ W_node + b_node      (exact f32 WMMA path)
// Block = 256 threads (8 waves); wave = 32 rows x 64 cols
// (2 row tiles x 4 col tiles of V_WMMA_F32_16X16X4_F32).
// B fragments batched into distinct registers before the WMMA burst.
// ---------------------------------------------------------------------------
__global__ void __launch_bounds__(256)
mpgc_gemm_out_kernel(const float* __restrict__ featS,
                     const float* __restrict__ featT,
                     const float* __restrict__ Wn,
                     const float* __restrict__ bias,
                     float* __restrict__ out)
{
    extern __shared__ float ldsW[];      // [OUTC][2*HID] (W_node transposed)
    const int tid = threadIdx.x;
    for (int i = tid; i < 2 * HID * OUTC; i += 256) {
        const int k = i / OUTC;
        const int c = i % OUTC;
        ldsW[c * (2 * HID) + k] = Wn[i];
    }
    __syncthreads();

    const int lane    = tid & 31;
    const int wave    = tid >> 5;
    const int rowBase = blockIdx.x * 256 + wave * 32;
    const int m       = lane & 15;
    const int khalf   = (lane >> 4) << 1;

    const float* rS[2];
    const float* rT[2];
#pragma unroll
    for (int rt = 0; rt < 2; ++rt) {
        int r = rowBase + rt * 16 + m;
        if (r >= N_NODES) r = N_NODES - 1;
        rS[rt] = featS + (size_t)r * HID;
        rT[rt] = featT + (size_t)r * HID;
    }

    v8f acc[2][4];
#pragma unroll
    for (int rt = 0; rt < 2; ++rt)
#pragma unroll
        for (int ct = 0; ct < 4; ++ct)
#pragma unroll
            for (int r = 0; r < 8; ++r) acc[rt][ct][r] = 0.0f;

    for (int k0 = 0; k0 < 2 * HID; k0 += 4) {
        v2f bf[4];
#pragma unroll
        for (int ct = 0; ct < 4; ++ct)
            bf[ct] = *(const v2f*)(ldsW + (ct * 16 + m) * (2 * HID) + k0 + khalf);
#pragma unroll
        for (int rt = 0; rt < 2; ++rt) {
            const float* src = (k0 < HID) ? (rS[rt] + k0) : (rT[rt] + (k0 - HID));
            const v2f a = *(const v2f*)(src + khalf);
#pragma unroll
            for (int ct = 0; ct < 4; ++ct) {
                acc[rt][ct] = __builtin_amdgcn_wmma_f32_16x16x4_f32(
                    false, a, false, bf[ct], (short)0, acc[rt][ct], false, false);
            }
        }
    }

    const int hi8 = (lane >> 4) << 3;
#pragma unroll
    for (int rt = 0; rt < 2; ++rt) {
#pragma unroll
        for (int ct = 0; ct < 4; ++ct) {
            const int   col = ct * 16 + m;
            const float bv  = bias[col];
#pragma unroll
            for (int r = 0; r < 8; ++r) {
                const int row = rowBase + rt * 16 + r + hi8;
                if (row < N_NODES) {
                    out[(size_t)row * OUTC + col] = acc[rt][ct][r] + bv;
                }
            }
        }
    }
}

// ---------------------------------------------------------------------------
// y[dst[e]] += w[e] * x[src[e]]  over 128 features.
// One wave per edge; lane handles 4 features (float4 gather + 4x
// global_atomic_add_f32). x and y are ~51 MB each -> L2 resident (192 MB L2).
// ---------------------------------------------------------------------------
__global__ void mpgc_spmm_kernel(const int* __restrict__ dst,
                                 const int* __restrict__ src,
                                 const float* __restrict__ w,
                                 const float* __restrict__ x,
                                 float* __restrict__ y,
                                 int nEdges)
{
    const int e = blockIdx.x * 8 + (threadIdx.x >> 5);
    if (e >= nEdges) return;
    const int lane = threadIdx.x & 31;

    const int   s  = src[e];
    const int   d  = dst[e];
    const float ww = w[e];

    const float4 v = ((const float4*)(x + (size_t)s * HID))[lane];
    float* yp = y + (size_t)d * HID + lane * 4;
    atomicAdd(yp + 0, ww * v.x);
    atomicAdd(yp + 1, ww * v.y);
    atomicAdd(yp + 2, ww * v.z);
    atomicAdd(yp + 3, ww * v.w);
}

// ---------------------------------------------------------------------------
// feat[i] += scale * nxt[i]; scale = w_s[h] (first half) / w_t[h] (second).
// ---------------------------------------------------------------------------
__global__ void mpgc_axpy_kernel(float* __restrict__ feat,
                                 const float* __restrict__ nxt,
                                 const float* __restrict__ ws,
                                 const float* __restrict__ wt,
                                 int h, size_t half)
{
    const float  a_s   = ws[h];
    const float  a_t   = wt[h];
    const size_t n4    = (2 * half) >> 2;
    const size_t half4 = half >> 2;
    for (size_t i = (size_t)blockIdx.x * blockDim.x + threadIdx.x; i < n4;
         i += (size_t)gridDim.x * blockDim.x) {
        const float  a = (i < half4) ? a_s : a_t;
        const float4 v = ((const float4*)nxt)[i];
        float4 f = ((float4*)feat)[i];
        f.x += a * v.x; f.y += a * v.y; f.z += a * v.z; f.w += a * v.w;
        ((float4*)feat)[i] = f;
    }
}

extern "C" void kernel_launch(void* const* d_in, const int* in_sizes, int n_in,
                              void* d_out, int out_size, void* d_ws, size_t ws_size,
                              hipStream_t stream) {
    const float* featureS = (const float*)d_in[0];
    const float* featureT = (const float*)d_in[1];
    const int*   erow     = (const int*)d_in[2];
    const int*   ecol     = (const int*)d_in[3];
    const float* ew       = (const float*)d_in[4];
    const float* Wsrc     = (const float*)d_in[5];
    const float* bsrc     = (const float*)d_in[6];
    const float* Wtgt     = (const float*)d_in[7];
    const float* btgt     = (const float*)d_in[8];
    const float* ws       = (const float*)d_in[9];
    const float* wt       = (const float*)d_in[10];
    const float* Wnode    = (const float*)d_in[11];
    const float* bnode    = (const float*)d_in[12];
    float*       out      = (float*)d_out;

    const size_t half = (size_t)N_NODES * HID;  // 12.8M floats per direction
    float* buf0 = (float*)d_ws;                 // curr (s | t)  102.4 MB
    float* buf1 = buf0 + 2 * half;              // next (s | t)  102.4 MB
    float* feat = buf1 + 2 * half;              // feat (s | t)  102.4 MB

    const dim3 blk(256);
    const int  gemmGrid = (N_NODES + 255) / 256;

    // x_s / x_t (+ feat init = w[0]*x) via bf16 split-3 WMMA
    const size_t ldsIn = (size_t)2 * FEAT * HID * sizeof(__bf16); // 128 KB
    mpgc_gemm_in_kernel<<<gemmGrid, blk, ldsIn, stream>>>(
        featureS, Wsrc, bsrc, ws, buf0, feat);
    mpgc_gemm_in_kernel<<<gemmGrid, blk, ldsIn, stream>>>(
        featureT, Wtgt, btgt, wt, buf0 + half, feat + half);

    // 3 hops of SpMM (both directions) + weighted accumulation
    float* curr = buf0;
    float* nxt  = buf1;
    const int spmmGrid = (N_EDGES + 7) / 8;
    for (int h = 1; h <= HOPS; ++h) {
        hipMemsetAsync(nxt, 0, 2 * half * sizeof(float), stream);
        // out[row] += w * curr_s[col]
        mpgc_spmm_kernel<<<spmmGrid, blk, 0, stream>>>(
            erow, ecol, ew, curr, nxt, N_EDGES);
        // out[col] += w * curr_t[row]   (adj^T)
        mpgc_spmm_kernel<<<spmmGrid, blk, 0, stream>>>(
            ecol, erow, ew, curr + half, nxt + half, N_EDGES);
        mpgc_axpy_kernel<<<4096, blk, 0, stream>>>(feat, nxt, ws, wt, h, half);
        float* tmp = curr; curr = nxt; nxt = tmp;
    }

    // out = [feat_s | feat_t] @ W_node + b_node via exact f32 WMMA
    const size_t ldsOut = (size_t)2 * HID * OUTC * sizeof(float); // 64 KB
    mpgc_gemm_out_kernel<<<gemmGrid, blk, ldsOut, stream>>>(
        feat, feat + half, Wnode, bnode, out);
}
